// Decoder_1202590842919
// MI455X (gfx1250) — matrix operand order repaired and checked by reading, not compiled, not run on hardware
//
#include <hip/hip_runtime.h>
#include <hip/hip_bf16.h>
#include <stdint.h>

typedef __attribute__((ext_vector_type(16))) __bf16 v16bf;
typedef __attribute__((ext_vector_type(8)))  __bf16 v8bf;
typedef __attribute__((ext_vector_type(8)))  float  v8f;

namespace {
constexpr int Bn   = 128;
constexpr int Tn   = 32;     // decode steps
constexpr int TFn  = 33;     // lang_indices inner dim
constexpr int Vn   = 30000;
constexpr int En   = 300;
constexpr int ENCn = 256;
constexpr int Dn   = 512;
constexpr int Gn   = 2048;   // 4*D
constexpr int KINn = 556;    // E + ENC
constexpr int KXn  = 576;    // padded to multiple of 32
constexpr int NB_FC = (Vn + 127) / 128;  // 235 vocab blocks of 128 (last partial: 48)
constexpr int LDS_K = 32;                // k-slice per stage
constexpr int LDS_PITCH = 40;            // 32 + 8 pad elems (80B rows: 16B aligned, conflict-free)
}

// ---------------- WMMA fragment helpers (bf16, 16x16x32) ----------------
// A (16xK tile): lane l holds row m0+(l&15); K = (l>>4)*8 + 0..7 and 16+(l>>4)*8 + 0..7
__device__ __forceinline__ v16bf load_a_frag(const __bf16* A, int lda, int m0, int k0, int lane) {
    const __bf16* p = A + (size_t)(m0 + (lane & 15)) * lda + k0 + ((lane >> 4) << 3);
    v8bf lo = *(const v8bf*)p;
    v8bf hi = *(const v8bf*)(p + 16);
    v16bf a;
#pragma unroll
    for (int i = 0; i < 8; ++i) { a[i] = lo[i]; a[8 + i] = hi[i]; }
    return a;
}

// B from row-major (N x K) weight matrix, computing A @ W^T:
// lane l holds output column n0+(l&15); K = (l>>4)*8 + 0..7 and 16+(l>>4)*8 + 0..7
__device__ __forceinline__ v16bf load_b_frag(const __bf16* W, int ldw, int n0, int k0, int lane) {
    const __bf16* p = W + (size_t)(n0 + (lane & 15)) * ldw + k0 + ((lane >> 4) << 3);
    v8bf lo = *(const v8bf*)p;
    v8bf hi = *(const v8bf*)(p + 16);
    v16bf b;
#pragma unroll
    for (int i = 0; i < 8; ++i) { b[i] = lo[i]; b[8 + i] = hi[i]; }
    return b;
}

#define WMMA_BF16(a, b, c) \
    __builtin_amdgcn_wmma_f32_16x16x32_bf16(false, (a), false, (b), (short)0, (c), false, false)

// Async global->LDS 16B copy (CDNA5 GLOBAL_LOAD_ASYNC_TO_LDS_B128, tracked by ASYNCcnt)
__device__ __forceinline__ void async_copy16(uint32_t lds_byte_addr, const void* gaddr) {
    asm volatile("global_load_async_to_lds_b128 %0, %1, off"
                 :: "v"(lds_byte_addr), "v"((uint64_t)(uintptr_t)gaddr) : "memory");
}

// ---------------- prep kernels ----------------
__global__ void cvt_bf16(const float* __restrict__ s, __bf16* __restrict__ d, long n) {
    long i = (long)blockIdx.x * blockDim.x + threadIdx.x;
    if (i < n) d[i] = (__bf16)s[i];
}

// W_ih (2048 x 556) -> bf16 padded (2048 x 576)
__global__ void build_wih(const float* __restrict__ W, __bf16* __restrict__ Wb, long n) {
    long i = (long)blockIdx.x * blockDim.x + threadIdx.x;
    if (i >= n) return;
    int col = (int)(i % KXn), row = (int)(i / KXn);
    float v = (col < KINn) ? W[(size_t)row * KINn + col] : 0.f;
    Wb[i] = (__bf16)v;
}

// X[b*T+t] = [ embedding(lang_indices[b][t]) (300) | enc[b] (256) | zeros (20) ]
__global__ void build_x(const int* __restrict__ lang_idx, const float* __restrict__ emb,
                        const float* __restrict__ enc, __bf16* __restrict__ Xb, long n) {
    long i = (long)blockIdx.x * blockDim.x + threadIdx.x;
    if (i >= n) return;
    int col = (int)(i % KXn);
    int row = (int)(i / KXn);
    int b = row / Tn, t = row % Tn;
    float v = 0.f;
    if (col < En) {
        int tok = lang_idx[b * TFn + t];
        v = emb[(size_t)tok * En + col];
    } else if (col < KINn) {
        v = enc[b * ENCn + (col - En)];
    }
    Xb[i] = (__bf16)v;
}

// h0 = enc @ W_init_h^T + b_init_h ; c0 likewise (small: plain fp32 dots)
__global__ void init_hc(const float* __restrict__ enc,
                        const float* __restrict__ Wh, const float* __restrict__ bh,
                        const float* __restrict__ Wc, const float* __restrict__ bc,
                        float* __restrict__ h, float* __restrict__ c, __bf16* __restrict__ hbf) {
    int i = blockIdx.x * blockDim.x + threadIdx.x;   // Bn*Dn threads
    int b = i >> 9, j = i & (Dn - 1);
    const float* e  = enc + b * ENCn;
    const float* wh = Wh + (size_t)j * ENCn;
    const float* wc = Wc + (size_t)j * ENCn;
    float sh = bh[j], sc = bc[j];
#pragma unroll 4
    for (int k = 0; k < ENCn; ++k) { float ev = e[k]; sh += ev * wh[k]; sc += ev * wc[k]; }
    h[i] = sh; c[i] = sc; hbf[i] = (__bf16)sh;
}

// ---------------- big GEMM: gatesX = X @ W_ih^T + b_ih + b_hh  (4096 x 2048, K=576)
__global__ void __launch_bounds__(256) gemm_gatesX(const __bf16* __restrict__ Xb,
                                                   const __bf16* __restrict__ Wih,
                                                   const float* __restrict__ bih,
                                                   const float* __restrict__ bhh,
                                                   float* __restrict__ gX) {
    const int lane = threadIdx.x & 31;
    const int wid  = (blockIdx.x * blockDim.x + threadIdx.x) >> 5;
    const int m0 = (wid >> 5) << 4;      // 256 M-tiles
    const int nc = (wid & 31) << 6;      // 32 chunks of 64 cols
    v8f a0 = {}, a1 = {}, a2 = {}, a3 = {};
    for (int k0 = 0; k0 < KXn; k0 += 32) {
        v16bf a  = load_a_frag(Xb, KXn, m0, k0, lane);
        v16bf b0 = load_b_frag(Wih, KXn, nc,      k0, lane);
        v16bf b1 = load_b_frag(Wih, KXn, nc + 16, k0, lane);
        v16bf b2 = load_b_frag(Wih, KXn, nc + 32, k0, lane);
        v16bf b3 = load_b_frag(Wih, KXn, nc + 48, k0, lane);
        a0 = WMMA_BF16(a, b0, a0);
        a1 = WMMA_BF16(a, b1, a1);
        a2 = WMMA_BF16(a, b2, a2);
        a3 = WMMA_BF16(a, b3, a3);
    }
    const int rbase = m0 + ((lane >> 4) << 3);
    const int cl = nc + (lane & 15);
#pragma unroll
    for (int r = 0; r < 8; ++r) {
        float* o = gX + (size_t)(rbase + r) * Gn;
        o[cl]      = a0[r] + bih[cl]      + bhh[cl];
        o[cl + 16] = a1[r] + bih[cl + 16] + bhh[cl + 16];
        o[cl + 32] = a2[r] + bih[cl + 32] + bhh[cl + 32];
        o[cl + 48] = a3[r] + bih[cl + 48] + bhh[cl + 48];
    }
}

// ---------------- per-step GEMM: gates = h @ W_hh^T + gatesX[:,t]  (128 x 2048, K=512)
__global__ void __launch_bounds__(256) gemm_gates_step(const __bf16* __restrict__ hbf,
                                                       const __bf16* __restrict__ Whh,
                                                       const float* __restrict__ gX,
                                                       float* __restrict__ gates, int t) {
    const int lane = threadIdx.x & 31;
    const int wid  = (blockIdx.x * blockDim.x + threadIdx.x) >> 5;
    const int m0 = (wid >> 5) << 4;      // 8 M-tiles
    const int nc = (wid & 31) << 6;      // 32 chunks of 64 cols
    const int rbase = m0 + ((lane >> 4) << 3);
    const int cl = nc + (lane & 15);
    v8f a0, a1, a2, a3;
#pragma unroll
    for (int r = 0; r < 8; ++r) {        // seed C operand with precomputed X-part + biases
        const float* gx = gX + (size_t)((rbase + r) * Tn + t) * Gn;
        a0[r] = gx[cl]; a1[r] = gx[cl + 16]; a2[r] = gx[cl + 32]; a3[r] = gx[cl + 48];
    }
    for (int k0 = 0; k0 < Dn; k0 += 32) {
        v16bf a  = load_a_frag(hbf, Dn, m0, k0, lane);
        v16bf b0 = load_b_frag(Whh, Dn, nc,      k0, lane);
        v16bf b1 = load_b_frag(Whh, Dn, nc + 16, k0, lane);
        v16bf b2 = load_b_frag(Whh, Dn, nc + 32, k0, lane);
        v16bf b3 = load_b_frag(Whh, Dn, nc + 48, k0, lane);
        a0 = WMMA_BF16(a, b0, a0);
        a1 = WMMA_BF16(a, b1, a1);
        a2 = WMMA_BF16(a, b2, a2);
        a3 = WMMA_BF16(a, b3, a3);
    }
#pragma unroll
    for (int r = 0; r < 8; ++r) {
        float* o = gates + (size_t)(rbase + r) * Gn;
        o[cl] = a0[r]; o[cl + 16] = a1[r]; o[cl + 32] = a2[r]; o[cl + 48] = a3[r];
    }
}

// ---------------- elementwise LSTM cell update ----------------
__global__ void lstm_step(const float* __restrict__ gates, const int* __restrict__ lang_len,
                          float* __restrict__ h, float* __restrict__ c,
                          __bf16* __restrict__ hbf, __bf16* __restrict__ hall,
                          int* __restrict__ actf, int t) {
    int i = blockIdx.x * blockDim.x + threadIdx.x;   // Bn*Dn threads
    int b = i >> 9, j = i & (Dn - 1);
    const float* g = gates + (size_t)b * Gn;
    float gi = g[j], gf = g[j + Dn], gg = g[j + 2 * Dn], go = g[j + 3 * Dn];
    float si = 1.f / (1.f + __expf(-gi));
    float sf = 1.f / (1.f + __expf(-gf));
    float so = 1.f / (1.f + __expf(-go));
    float c_new = sf * c[i] + si * tanhf(gg);
    float h_new = so * tanhf(c_new);
    bool act = t < (lang_len[b] - 1);
    float h_keep = act ? h_new : h[i];
    float c_keep = act ? c_new : c[i];
    h[i] = h_keep; c[i] = c_keep;
    hbf[i] = (__bf16)h_keep;                                 // carried state feeds next step
    hall[(size_t)(b * Tn + t) * Dn + j] = (__bf16)h_new;     // pre-mask state feeds preds
    if (j == 0) actf[b * Tn + t] = act ? 1 : 0;
}

// ---------------- vocab projection: out = h_all @ W_fc^T + b_fc (masked) ----------------
// (4096 x 30000, K=512). 8 waves cooperate on a 128x128 block; A/B k-slices are
// double-buffered in LDS via async-to-LDS copies (ASYNCcnt), WMMAs read from LDS,
// outputs stream to HBM with non-temporal stores so W_fc stays L2-resident.
__global__ void __launch_bounds__(256) gemm_fc(const __bf16* __restrict__ hall,
                                               const __bf16* __restrict__ Wfc,
                                               const float* __restrict__ bfc,
                                               const int* __restrict__ actf,
                                               float* __restrict__ out) {
    __shared__ __bf16 As[2][128 * LDS_PITCH];
    __shared__ __bf16 Bs[2][128 * LDS_PITCH];

    const int tid  = threadIdx.x;
    const int lane = tid & 31;
    const int wave = tid >> 5;
    const int m0 = (blockIdx.x / NB_FC) * 128;
    const int n0 = (blockIdx.x % NB_FC) * 128;

    // this thread's two 16B chunks per 128x32 stage: chunk = tid*2+q, row = chunk/4, kc = (chunk%4)*8
    const int ch0  = tid * 2;
    const int rowA = ch0 >> 2;
    const int kc   = (ch0 & 3) * 8;          // element offset within 32-k slice (q=0; q=1 -> +8)
    const __bf16* gA = hall + (size_t)(m0 + rowA) * Dn + kc;
    const __bf16* gB = Wfc  + (size_t)(n0 + rowA) * Dn + kc;  // tail block may over-read into ws (discarded)
    const uint32_t lA = (uint32_t)(uintptr_t)&As[0][rowA * LDS_PITCH + kc];
    const uint32_t lB = (uint32_t)(uintptr_t)&Bs[0][rowA * LDS_PITCH + kc];
    const uint32_t bufStride = (uint32_t)(128 * LDS_PITCH * sizeof(__bf16));

    v8f acc[8];
#pragma unroll
    for (int s = 0; s < 8; ++s) acc[s] = (v8f){};

    // prologue: stage k-slice 0 into buffer 0
    async_copy16(lA,      gA);
    async_copy16(lA + 16, gA + 8);
    async_copy16(lB,      gB);
    async_copy16(lB + 16, gB + 8);

    const int m_local = wave * 16;
#pragma unroll 1
    for (int ki = 0; ki < Dn / LDS_K; ++ki) {
        const int cur = ki & 1;
        if (ki + 1 < Dn / LDS_K) {
            const uint32_t lo = (cur ^ 1) * bufStride;
            const __bf16* a = gA + (ki + 1) * LDS_K;
            const __bf16* b = gB + (ki + 1) * LDS_K;
            async_copy16(lA + lo,      a);
            async_copy16(lA + lo + 16, a + 8);
            async_copy16(lB + lo,      b);
            async_copy16(lB + lo + 16, b + 8);
            asm volatile("s_wait_asynccnt 0x4" ::: "memory");  // slice ki complete (in-order)
        } else {
            asm volatile("s_wait_asynccnt 0x0" ::: "memory");
        }
        __syncthreads();
        const __bf16* Ab = &As[cur][0];
        const __bf16* Bb = &Bs[cur][0];
        v16bf af = load_a_frag(Ab, LDS_PITCH, m_local, 0, lane);
#pragma unroll
        for (int s = 0; s < 8; ++s) {
            v16bf bf = load_b_frag(Bb, LDS_PITCH, s * 16, 0, lane);
            acc[s] = WMMA_BF16(af, bf, acc[s]);
        }
        __syncthreads();
    }

    // store 128x128 block, masked + bias, streaming (non-temporal) to protect L2
    const int rbase = m_local + ((lane >> 4) << 3);
    const int cl = lane & 15;
#pragma unroll
    for (int r = 0; r < 8; ++r) {
        const int grow = m0 + rbase + r;
        const bool act = actf[grow] != 0;
        float* o = out + (size_t)grow * Vn;
#pragma unroll
        for (int s = 0; s < 8; ++s) {
            const int col = n0 + s * 16 + cl;
            if (col < Vn) {
                float v = act ? (acc[s][r] + bfc[col]) : 0.f;
                __builtin_nontemporal_store(v, &o[col]);
            }
        }
    }
}

// ---------------- host launch ----------------
extern "C" void kernel_launch(void* const* d_in, const int* in_sizes, int n_in,
                              void* d_out, int out_size, void* d_ws, size_t ws_size,
                              hipStream_t stream) {
    const float* ref_obj  = (const float*)d_in[0];
    const int*   lang_idx = (const int*)d_in[1];
    const int*   lang_len = (const int*)d_in[2];
    const float* emb      = (const float*)d_in[3];
    const float* W_ih     = (const float*)d_in[4];
    const float* b_ih     = (const float*)d_in[5];
    const float* W_hh     = (const float*)d_in[6];
    const float* b_hh     = (const float*)d_in[7];
    const float* W_init_h = (const float*)d_in[8];
    const float* b_init_h = (const float*)d_in[9];
    const float* W_init_c = (const float*)d_in[10];
    const float* b_init_c = (const float*)d_in[11];
    const float* W_fc     = (const float*)d_in[12];
    const float* b_fc     = (const float*)d_in[13];
    float* out = (float*)d_out;
    (void)in_sizes; (void)n_in; (void)out_size; (void)ws_size;

    char* p = (char*)d_ws;
    auto take = [&](size_t bytes) { char* r = p; p += (bytes + 255) & ~(size_t)255; return r; };
    __bf16* Xb     = (__bf16*)take((size_t)Bn * Tn * KXn * 2);
    __bf16* Wihb   = (__bf16*)take((size_t)Gn * KXn * 2);
    __bf16* Whhb   = (__bf16*)take((size_t)Gn * Dn * 2);
    __bf16* Wfcb   = (__bf16*)take((size_t)Vn * Dn * 2);
    float*  gatesX = (float*)take((size_t)Bn * Tn * Gn * 4);   // directly after Wfcb: tail over-reads land here
    float*  gates  = (float*)take((size_t)Bn * Gn * 4);
    float*  hs     = (float*)take((size_t)Bn * Dn * 4);
    float*  cs     = (float*)take((size_t)Bn * Dn * 4);
    __bf16* hbf    = (__bf16*)take((size_t)Bn * Dn * 2);
    __bf16* hall   = (__bf16*)take((size_t)Bn * Tn * Dn * 2);
    int*    actf   = (int*)take((size_t)Bn * Tn * 4);

    // weight quantization + input assembly (all parallel, no deps between them)
    { long n = (long)Gn * Dn;  cvt_bf16<<<(unsigned)((n + 255) / 256), 256, 0, stream>>>(W_hh, Whhb, n); }
    { long n = (long)Vn * Dn;  cvt_bf16<<<(unsigned)((n + 255) / 256), 256, 0, stream>>>(W_fc, Wfcb, n); }
    { long n = (long)Gn * KXn; build_wih<<<(unsigned)((n + 255) / 256), 256, 0, stream>>>(W_ih, Wihb, n); }
    { long n = (long)Bn * Tn * KXn;
      build_x<<<(unsigned)((n + 255) / 256), 256, 0, stream>>>(lang_idx, emb, ref_obj, Xb, n); }
    init_hc<<<(Bn * Dn) / 256, 256, 0, stream>>>(ref_obj, W_init_h, b_init_h, W_init_c, b_init_c,
                                                 hs, cs, hbf);

    // hoisted input-side gate GEMM for all 4096 (b,t) rows
    gemm_gatesX<<<(256 * 32) / 8, 256, 0, stream>>>(Xb, Wihb, b_ih, b_hh, gatesX);

    // sequential recurrence (tiny per-step work)
    for (int t = 0; t < Tn; ++t) {
        gemm_gates_step<<<(8 * 32) / 8, 256, 0, stream>>>(hbf, Whhb, gatesX, gates, t);
        lstm_step<<<(Bn * Dn) / 256, 256, 0, stream>>>(gates, lang_len, hs, cs, hbf, hall, actf, t);
    }

    // one massive LDS-staged vocab-projection GEMM (store-bandwidth bound)
    gemm_fc<<<32 * NB_FC, 256, 0, stream>>>(hall, Wfcb, b_fc, actf, out);
}
